// NeuralRDE_79173427134598
// MI455X (gfx1250) — matrix operand order repaired and checked by reading, not compiled, not run on hardware
//
#include <hip/hip_runtime.h>
#include <hip/hip_bf16.h>

// Problem constants (from reference)
#define BATCH 1024
#define DD    6
#define HH    128
#define VWW   256
#define LSS   22
#define LL    21          // LS-1
#define NII   64
#define NSTEP 64
#define AHH   (HH*LL)     // 2688
#define BT    16          // batch rows per block
#define NTHREADS 256      // 8 waves (wave32)
#define NWAVES   8
#define STAGE_BYTES (2 * NWAVES * BT * VWW * 2)   // 128 KB dynamic LDS

typedef __bf16 bf16;
typedef __attribute__((ext_vector_type(16))) __bf16 v16bf;
typedef __attribute__((ext_vector_type(8)))  __bf16 v8bf;
typedef __attribute__((ext_vector_type(8)))  float  v8f;

__device__ __forceinline__ v8f wmma_bf16(v16bf a, v16bf b, v8f c) {
  // (neg_a, A, neg_b, B, c_mod, C, reuse_a, reuse_b)
  return __builtin_amdgcn_wmma_f32_16x16x32_bf16(false, a, false, b, (short)0, c,
                                                 false, false);
}

// 32-bit LDS byte offset (addrspace 3) of a shared-memory object.
__device__ __forceinline__ unsigned lds_off(const void* p) {
  return (unsigned)(unsigned long long)
      (__attribute__((address_space(3))) const char*)p;
}

// ASYNCcnt-tracked wait: all async global<->LDS copies complete.
__device__ __forceinline__ void wait_async0() {
  asm volatile("s_wait_asynccnt 0x0" ::: "memory");
}

// One lane-parallel async b128 copy: LDS[l+IMM] = MEM[g+IMM] (16B per lane).
#define ASYNC_CP16(l, g, IMM)                                                \
  asm volatile("global_load_async_to_lds_b128 %0, %1, off offset:" #IMM      \
               :: "v"(l), "v"(g) : "memory")

// Async-stage one 16x256 bf16 weight tile (rows n0..n0+15 of W, ldk=256)
// into `stage` (row-major 16x256). Row stride is 512B in both global and LDS,
// so one base + 16 instruction offsets covers the tile. 8 KB total.
__device__ __forceinline__ void async_load_tile(const bf16* W, int n0,
                                                bf16* stage, int lane) {
  const bf16* g = W + (size_t)n0 * VWW + lane * 8;   // 16B per lane per row
  unsigned    l = lds_off(stage) + lane * 16;
  ASYNC_CP16(l, g, 0);    ASYNC_CP16(l, g, 512);  ASYNC_CP16(l, g, 1024);
  ASYNC_CP16(l, g, 1536); ASYNC_CP16(l, g, 2048); ASYNC_CP16(l, g, 2560);
  ASYNC_CP16(l, g, 3072); ASYNC_CP16(l, g, 3584); ASYNC_CP16(l, g, 4096);
  ASYNC_CP16(l, g, 4608); ASYNC_CP16(l, g, 5120); ASYNC_CP16(l, g, 5632);
  ASYNC_CP16(l, g, 6144); ASYNC_CP16(l, g, 6656); ASYNC_CP16(l, g, 7168);
  ASYNC_CP16(l, g, 7680);
}

// A fragment (16x32 bf16, row-major source with leading dim ld).
// Lane l holds row m=l&15; K contiguous at k0 + 16*(l>>4).
__device__ __forceinline__ v16bf load_a_frag(const bf16* m0, int ld, int lane, int k0) {
  int m  = lane & 15;
  int ks = (lane >> 4) << 4;          // 0 or 16
  const bf16* p = m0 + m * ld + k0;
  union { v16bf v; v8bf h[2]; } u;
  u.h[0] = *(const v8bf*)(p + ks);
  u.h[1] = *(const v8bf*)(p + 8 + ks);
  return u.v;
}

// B fragment (32x16 bf16) for out = act * W^T: B[k][n] = W[n0+n][k], W row-major, ldk.
// Lane l holds column n=l&15, K contiguous at k0 + 16*(l>>4).
__device__ __forceinline__ v16bf load_b_frag(const bf16* W, int ldk, int n0, int lane, int k0) {
  int n  = lane & 15;
  int ks = (lane >> 4) << 4;          // 0 or 16
  return *(const v16bf*)(W + (size_t)(n0 + n) * ldk + k0 + ks);
}

// One vector-field evaluation for a 16-row batch tile, fully in LDS.
__device__ __forceinline__ void eval_func(
    int tid, int lane, int w, int b0, float t,
    const float* tsL, const float* logsig,
    const float* b_vf1, const float* b_vf2, const float* bm,
    const bf16* wv1, const bf16* wv2, const bf16* wm,
    const bf16 (*ybf)[HH], bf16 (*hbf)[VWW], bf16 (*vbf)[VWW],
    float (*lsw)[LL], float (*dst)[HH], bf16* bstage,
    int* s_idx, float* s_invw)
{
  if (tid == 0) {
    // searchsorted(ts, t) clipped to [1, NI]
    int idx = 0;
    while (idx <= NII && tsL[idx] < t) ++idx;
    if (idx < 1)   idx = 1;
    if (idx > NII) idx = NII;
    *s_idx  = idx;
    *s_invw = 1.0f / (tsL[idx] - tsL[idx - 1]);
  }
  __syncthreads();  // publishes caller-written ybf and s_idx/s_invw

  const int   idx  = *s_idx;
  const float invw = *s_invw;
  // ls tile, width folded in: lsw[b][l] = logsig[b0+b, idx-1, l+1] / width
  for (int i = tid; i < BT * LL; i += NTHREADS) {
    int bb = i / LL, l = i % LL;
    lsw[bb][l] = logsig[((size_t)(b0 + bb) * NII + (idx - 1)) * LSS + (l + 1)] * invw;
  }
  // zero accumulation target
  for (int i = tid; i < BT * HH; i += NTHREADS)
    dst[i >> 7][i & (HH - 1)] = 0.0f;
  __syncthreads();

  // ---- GEMM1: h = relu(y @ W_vf1^T + b_vf1), 16x256, K=128 ----
  {
    const int n0 = w * 32;
    v8f acc0 = {}, acc1 = {};
#pragma unroll
    for (int kc = 0; kc < HH; kc += 32) {
      v16bf a  = load_a_frag(&ybf[0][0], HH, lane, kc);
      v16bf bA = load_b_frag(wv1, HH, n0,      lane, kc);
      v16bf bB = load_b_frag(wv1, HH, n0 + 16, lane, kc);
      acc0 = wmma_bf16(a, bA, acc0);
      acc1 = wmma_bf16(a, bB, acc1);
    }
    int n = lane & 15, mb = (lane >> 4) << 3;
    float bias0 = b_vf1[n0 + n], bias1 = b_vf1[n0 + 16 + n];
#pragma unroll
    for (int r = 0; r < 8; ++r) {
      int m = mb + r;
      hbf[m][n0 + n]      = (bf16)fmaxf(acc0[r] + bias0, 0.0f);
      hbf[m][n0 + 16 + n] = (bf16)fmaxf(acc1[r] + bias1, 0.0f);
    }
  }
  __syncthreads();

  // ---- GEMM2: v = tanh(h @ W_vf2^T + b_vf2), 16x256, K=256 ----
  {
    const int n0 = w * 32;
    v8f acc0 = {}, acc1 = {};
#pragma unroll
    for (int kc = 0; kc < VWW; kc += 32) {
      v16bf a  = load_a_frag(&hbf[0][0], VWW, lane, kc);
      v16bf bA = load_b_frag(wv2, VWW, n0,      lane, kc);
      v16bf bB = load_b_frag(wv2, VWW, n0 + 16, lane, kc);
      acc0 = wmma_bf16(a, bA, acc0);
      acc1 = wmma_bf16(a, bB, acc1);
    }
    int n = lane & 15, mb = (lane >> 4) << 3;
    float bias0 = b_vf2[n0 + n], bias1 = b_vf2[n0 + 16 + n];
#pragma unroll
    for (int r = 0; r < 8; ++r) {
      int m = mb + r;
      vbf[m][n0 + n]      = (bf16)tanhf(acc0[r] + bias0);
      vbf[m][n0 + 16 + n] = (bf16)tanhf(acc1[r] + bias1);
    }
  }
  __syncthreads();

  // ---- GEMM3 + fused contraction: A = v @ Wm^T + bm (16x2688),
  //      out[b,h'] += A[b, h'*21+l] * ls[b,l]/width.
  // Wave w owns flat columns [w*336, w*336+336) == heads [16w, 16w+16).
  // Wm tiles are async-staged global->LDS (double buffered, ASYNCcnt),
  // A fragments are hoisted into registers for the whole tile loop.
  {
    bf16* st0 = bstage + (size_t)(0 * NWAVES + w) * BT * VWW;
    bf16* st1 = bstage + (size_t)(1 * NWAVES + w) * BT * VWW;

    // kick off tile 0 before touching anything else
    async_load_tile(wm, w * (16 * LL), st0, lane);

    v16bf afr[8];
#pragma unroll
    for (int kc8 = 0; kc8 < 8; ++kc8)
      afr[kc8] = load_a_frag(&vbf[0][0], VWW, lane, kc8 * 32);

    const int n   = lane & 15;
    const int mb  = (lane >> 4) << 3;
    const int ks  = (lane >> 4) << 4;

#pragma unroll 1
    for (int i2 = 0; i2 < LL; ++i2) {
      bf16* cur = (i2 & 1) ? st1 : st0;
      bf16* nxt = (i2 & 1) ? st0 : st1;
      wait_async0();                        // tile i2 resident in LDS
      if (i2 + 1 < LL)                      // overlap next tile with compute
        async_load_tile(wm, w * (16 * LL) + (i2 + 1) * 16, nxt, lane);

      v8f acc = {};
#pragma unroll
      for (int kc8 = 0; kc8 < 8; ++kc8) {
        v16bf bfrag = *(const v16bf*)(cur + n * VWW + kc8 * 32 + ks);
        acc = wmma_bf16(afr[kc8], bfrag, acc);
      }

      const int n0   = w * (16 * LL) + i2 * 16;
      const int flat = n0 + n;
      const int hp   = flat / LL, lc = flat % LL;
      float bmv = bm[flat];
#pragma unroll
      for (int r = 0; r < 8; ++r) {
        int m = mb + r;
        atomicAdd(&dst[m][hp], (acc[r] + bmv) * lsw[m][lc]);
      }
    }
  }
  __syncthreads();
}

__global__ void __launch_bounds__(NTHREADS)
NeuralRDE_persistent_kernel(
    const float* ts, const float* logsig, const float* x0,
    const float* b_vf1, const float* b_vf2, const float* bm,
    const float* W1, const float* b1, const float* W2, const float* b2,
    const bf16* wv1, const bf16* wv2, const bf16* wm,
    float* out)
{
  __shared__ float yf [BT][HH];
  __shared__ float k1f[BT][HH];
  __shared__ float k2f[BT][HH];
  __shared__ bf16  ybf[BT][HH];
  __shared__ bf16  hbf[BT][VWW];
  __shared__ bf16  vbf[BT][VWW];
  __shared__ float lsw[BT][LL];
  __shared__ float tsL[NII + 1];
  __shared__ float logit[BT][10];
  __shared__ int   s_idx;
  __shared__ float s_invw;
  extern __shared__ __attribute__((aligned(64))) char dynsmem[];  // 128 KB stage
  bf16* bstage = (bf16*)dynsmem;

  const int tid  = threadIdx.x;
  const int lane = tid & 31;
  const int w    = tid >> 5;
  const int b0   = blockIdx.x * BT;

  for (int i = tid; i <= NII; i += NTHREADS) tsL[i] = ts[i];

  // y0 = x0 @ W1^T + b1  (K=6, scalar VALU)
  for (int i = tid; i < BT * HH; i += NTHREADS) {
    int bb = i >> 7, hh = i & (HH - 1);
    float s = b1[hh];
#pragma unroll
    for (int d = 0; d < DD; ++d)
      s += x0[(size_t)(b0 + bb) * DD + d] * W1[hh * DD + d];
    yf[bb][hh] = s;
  }
  __syncthreads();

  const float t0 = tsL[0];
  const float dt = (tsL[NII] - tsL[0]) / (float)NSTEP;

  for (int j = 0; j < NSTEP; ++j) {
    const float t = t0 + dt * (float)j;

    // k1 = func(t, y)
    for (int i = tid; i < BT * HH; i += NTHREADS) {
      int bb = i >> 7, hh = i & (HH - 1);
      ybf[bb][hh] = (bf16)yf[bb][hh];
    }
    eval_func(tid, lane, w, b0, t, tsL, logsig, b_vf1, b_vf2, bm,
              wv1, wv2, wm, ybf, hbf, vbf, lsw, k1f, bstage, &s_idx, &s_invw);

    // k2 = func(t+dt, y + dt*k1)
    for (int i = tid; i < BT * HH; i += NTHREADS) {
      int bb = i >> 7, hh = i & (HH - 1);
      ybf[bb][hh] = (bf16)(yf[bb][hh] + dt * k1f[bb][hh]);
    }
    eval_func(tid, lane, w, b0, t + dt, tsL, logsig, b_vf1, b_vf2, bm,
              wv1, wv2, wm, ybf, hbf, vbf, lsw, k2f, bstage, &s_idx, &s_invw);

    // y += 0.5*dt*(k1+k2)
    for (int i = tid; i < BT * HH; i += NTHREADS) {
      int bb = i >> 7, hh = i & (HH - 1);
      yf[bb][hh] += 0.5f * dt * (k1f[bb][hh] + k2f[bb][hh]);
    }
    __syncthreads();
  }

  // logits = yT @ W2^T + b2 ; softmax
  for (int i = tid; i < BT * 10; i += NTHREADS) {
    int bb = i / 10, c = i % 10;
    float s = b2[c];
    for (int hh = 0; hh < HH; ++hh) s += yf[bb][hh] * W2[c * HH + hh];
    logit[bb][c] = s;
  }
  __syncthreads();
  if (tid < BT) {
    int bb = tid;
    float mx = logit[bb][0];
#pragma unroll
    for (int c = 1; c < 10; ++c) mx = fmaxf(mx, logit[bb][c]);
    float e[10], se = 0.0f;
#pragma unroll
    for (int c = 0; c < 10; ++c) { e[c] = __expf(logit[bb][c] - mx); se += e[c]; }
    float inv = 1.0f / se;
#pragma unroll
    for (int c = 0; c < 10; ++c) out[(size_t)(b0 + bb) * 10 + c] = e[c] * inv;
  }
}

// Convert the three GEMM weight matrices fp32 -> bf16 into workspace (L2-resident).
__global__ void prep_weights_kernel(const float* a, const float* b, const float* c,
                                    bf16* oa, bf16* ob, bf16* oc)
{
  int i = blockIdx.x * blockDim.x + threadIdx.x;
  if (i < VWW * HH)  oa[i] = (bf16)a[i];
  if (i < VWW * VWW) ob[i] = (bf16)b[i];
  if (i < AHH * VWW) oc[i] = (bf16)c[i];
}

extern "C" void kernel_launch(void* const* d_in, const int* in_sizes, int n_in,
                              void* d_out, int out_size, void* d_ws, size_t ws_size,
                              hipStream_t stream)
{
  (void)in_sizes; (void)n_in; (void)out_size; (void)ws_size;
  const float* ts     = (const float*)d_in[0];
  const float* logsig = (const float*)d_in[1];
  const float* x0     = (const float*)d_in[2];
  const float* W_vf1  = (const float*)d_in[3];
  const float* b_vf1  = (const float*)d_in[4];
  const float* W_vf2  = (const float*)d_in[5];
  const float* b_vf2  = (const float*)d_in[6];
  const float* Wm     = (const float*)d_in[7];
  const float* bm     = (const float*)d_in[8];
  const float* W1     = (const float*)d_in[9];
  const float* b1     = (const float*)d_in[10];
  const float* W2     = (const float*)d_in[11];
  const float* b2     = (const float*)d_in[12];

  bf16* wsb = (bf16*)d_ws;
  bf16* wv1 = wsb;                        // 256*128
  bf16* wv2 = wsb + VWW * HH;             // 256*256
  bf16* wm  = wsb + VWW * HH + VWW * VWW; // 2688*256

  const int nconv = AHH * VWW;            // largest matrix drives the grid
  prep_weights_kernel<<<(nconv + 255) / 256, 256, 0, stream>>>(
      W_vf1, W_vf2, Wm, wv1, wv2, wm);

  // 320 KB LDS per WGP on CDNA5: opt in to a 128 KB dynamic stage buffer.
  hipFuncSetAttribute((const void*)NeuralRDE_persistent_kernel,
                      hipFuncAttributeMaxDynamicSharedMemorySize, STAGE_BYTES);

  NeuralRDE_persistent_kernel<<<BATCH / BT, NTHREADS, STAGE_BYTES, stream>>>(
      ts, logsig, x0, b_vf1, b_vf2, bm, W1, b1, W2, b2,
      wv1, wv2, wm, (float*)d_out);
}
